// EdgeHGNN_5763846111292
// MI455X (gfx1250) — matrix operand order repaired and checked by reading, not compiled, not run on hardware
//
#include <hip/hip_runtime.h>
#include <hip/hip_bf16.h>

// ---------------- problem constants ----------------
#define NEDGE   100000
#define LATDIM  64
#define HYPERN  128
#define NUSER   100000
#define NITEM   100000
#define NTYPE   8

typedef __attribute__((ext_vector_type(16))) _Float16 v16h;
typedef __attribute__((ext_vector_type(8)))  float    v8f;

__device__ __forceinline__ v8f wmma_f16f32(v16h a, v16h b, v8f c) {
    // v_wmma_f32_16x16x32_f16: D = A(16x32) * B(32x16) + C(16x16 f32)
    return __builtin_amdgcn_wmma_f32_16x16x32_f16(
        false, a, false, b, (short)0, c, false, false);
}

// ---- WMMA operand swizzle (CDNA5 wave32 layouts, CDNA5 ISA §7.12.2) ----
// Fragment-linear storage: frag[lane*16 + elem]; a v16h load per lane = 2x ds_load_b128.
// A 16x32 f16: lane = m + 16*half, k = 2*(j&3) + 16*(j>>2) + 8*half, elem = 2j+(k&1)
__device__ __forceinline__ int a_pos(int m, int k) {
    int half = (k >> 3) & 1;
    int j    = ((k >> 4) << 2) | ((k & 7) >> 1);
    return ((m + (half << 4)) << 4) + 2 * j + (k & 1);
}
// B 32x16 f16: lane = n + 16*half, k = 2*(j&3) + 16*(j>>2) + 8*half, elem = 2j+(k&1)
__device__ __forceinline__ int b_pos(int k, int n) {
    int lane = n + (((k >> 3) & 1) << 4);
    int j    = ((k >> 4) << 2) | ((k & 7) >> 1);
    return (lane << 4) + 2 * j + (k & 1);
}

// A 16x32 f16 built on the fly from row-major f32 source A[m][k], row stride ld.
__device__ __forceinline__ v16h frag_a_f32(const float* base, int ld, int lane) {
    union { v16h v; _Float16 e[16]; } x;
    const int m  = lane & 15;
    const int h8 = (lane >> 4) << 3;
    const float* p = base + m * ld;
#pragma unroll
    for (int j = 0; j < 8; ++j) {
        int k = ((j & 3) << 1) + ((j >> 2) << 4) + h8; // even
        float2 f = *(const float2*)(p + k);
        x.e[2 * j]     = (_Float16)f.x;
        x.e[2 * j + 1] = (_Float16)f.y;
    }
    return x.v;
}

// ---- CDNA5 async global->LDS copy (16B per lane), tracked by ASYNCcnt ----
__device__ __forceinline__ void async_copy16(void* lds_dst, const void* gsrc) {
    unsigned l = (unsigned)(uintptr_t)lds_dst;
    unsigned long long g = (unsigned long long)(uintptr_t)gsrc;
    asm volatile("global_load_async_to_lds_b128 %0, %1, off"
                 :: "v"(l), "v"(g) : "memory");
}
__device__ __forceinline__ void async_wait0() {
    asm volatile("s_wait_asynccnt 0x0" ::: "memory");
}

__device__ __forceinline__ float leaky05(float x) { return x > 0.0f ? x : 0.5f * x; }

// ---------------- kernels ----------------

__global__ void k_zero_f32(float* __restrict__ p, int n) {
    int t = blockIdx.x * blockDim.x + threadIdx.x;
    if (t < n) p[t] = 0.0f;
}

// hyper f32 [e][h] -> hyB: per 32-edge slab, B-fragment order [slab][nt(8)][lane][16]
__global__ void k_cvt_hyB(const float* __restrict__ hyp, _Float16* __restrict__ hyB, int n) {
    int t = blockIdx.x * blockDim.x + threadIdx.x;
    if (t < n) {
        int e = t >> 7, h = t & 127;
        int slab = e >> 5, k = e & 31;
        int nt = h >> 4, nn = h & 15;
        hyB[slab * 4096 + nt * 512 + b_pos(k, nn)] = (_Float16)hyp[t];
    }
}

// edgeEmbeds0[e,:] = (uEmbeds[src[e],:] @ edgeTrans[val[e]]) * iEmbeds[tgt[e],:]
// writes f32 linear + f16 in A-fragment slab order eeHf[slab][mt(4)][lane][16]
__global__ void k_edge_init(const float* __restrict__ uE, const float* __restrict__ iE,
                            const float* __restrict__ eT,
                            const int* __restrict__ src, const int* __restrict__ tgt,
                            const int* __restrict__ val,
                            float* __restrict__ eeF, _Float16* __restrict__ eeHf) {
    extern __shared__ float smem[];
    float* sT   = smem;                            // 8*64*64 floats (128 KB)
    float* sSrc = smem + NTYPE * LATDIM * LATDIM;  // 8 waves * 64 floats

    const int tid = threadIdx.x;
    for (int i = tid; i < NTYPE * LATDIM * LATDIM; i += blockDim.x) sT[i] = eT[i];
    __syncthreads();

    const int lane = tid & 31;
    const int w    = tid >> 5;
    float* mySrc = sSrc + w * LATDIM;

    for (int e = blockIdx.x * 8 + w; e < NEDGE; e += gridDim.x * 8) {
        const int s = src[e], tg = tgt[e], v = val[e];
        mySrc[lane]      = uE[s * LATDIM + lane];
        mySrc[lane + 32] = uE[s * LATDIM + lane + 32];
        const float* T = sT + v * LATDIM * LATDIM;
        float a0 = 0.0f, a1 = 0.0f;
#pragma unroll 16
        for (int d = 0; d < LATDIM; ++d) {
            float sd = mySrc[d]; // LDS broadcast
            a0 += sd * T[d * LATDIM + lane];
            a1 += sd * T[d * LATDIM + lane + 32];
        }
        float r0 = a0 * iE[tg * LATDIM + lane];
        float r1 = a1 * iE[tg * LATDIM + lane + 32];
        const int b = e * LATDIM;
        eeF[b + lane]      = r0;
        eeF[b + lane + 32] = r1;
        const int slab = e >> 5, k = e & 31;
        _Float16* eh = eeHf + slab * 2048;
        const int d0 = lane, d1 = lane + 32;
        eh[(d0 >> 4) * 512 + a_pos(d0 & 15, k)] = (_Float16)r0;
        eh[(d1 >> 4) * 512 + a_pos(d1 & 15, k)] = (_Float16)r1;
    }
}

// acc[d,h] += sum_e ee[e,d] * hyper[e,h]  — split-K WMMA over 32-edge slabs.
// Fragments are pre-swizzled in global memory; LDS staging is a raw async copy.
__global__ void k_gemm1_accum(const _Float16* __restrict__ eeHf,
                              const _Float16* __restrict__ hyB,
                              float* __restrict__ acc) {
    __shared__ __align__(32) _Float16 sE[2048]; // 4 A-frags (4 KB)
    __shared__ __align__(32) _Float16 sH[4096]; // 8 B-frags (8 KB)

    const int tid  = threadIdx.x;
    const int lane = tid & 31;
    const int w    = tid >> 5;

    v8f c0 = {}, c1 = {}, c2 = {}, c3 = {};

    const int nslab = NEDGE / 32; // 3125
    for (int s = blockIdx.x; s < nslab; s += gridDim.x) {
        __syncthreads(); // previous iteration's reads complete
        const _Float16* gE = eeHf + (size_t)s * 2048;
        const _Float16* gH = hyB  + (size_t)s * 4096;
        async_copy16(sE + tid * 8,        gE + tid * 8);
        async_copy16(sH + tid * 8,        gH + tid * 8);
        async_copy16(sH + 2048 + tid * 8, gH + 2048 + tid * 8);
        async_wait0();
        __syncthreads();

        v16h b = ((const v16h*)sH)[w * 32 + lane];      // 2x ds_load_b128
        v16h a;
        a = ((const v16h*)sE)[0 * 32 + lane];  c0 = wmma_f16f32(a, b, c0);
        a = ((const v16h*)sE)[1 * 32 + lane];  c1 = wmma_f16f32(a, b, c1);
        a = ((const v16h*)sE)[2 * 32 + lane];  c2 = wmma_f16f32(a, b, c2);
        a = ((const v16h*)sE)[3 * 32 + lane];  c3 = wmma_f16f32(a, b, c3);
    }

    const int n  = (w << 4) + (lane & 15);
    const int mh = (lane >> 4) << 3;
#pragma unroll
    for (int r = 0; r < 8; ++r) {
        atomicAdd(acc + (0 * 16 + mh + r) * HYPERN + n, c0[r]);
        atomicAdd(acc + (1 * 16 + mh + r) * HYPERN + n, c1[r]);
        atomicAdd(acc + (2 * 16 + mh + r) * HYPERN + n, c2[r]);
        atomicAdd(acc + (3 * 16 + mh + r) * HYPERN + n, c3[r]);
    }
}

// h1f = f16(leaky(acc[d,h])) stored as 16 B-fragments [ks(4)][nt(4)][lane][16]
__global__ void k_act_h1(const float* __restrict__ acc, _Float16* __restrict__ h1f) {
    int t = blockIdx.x * blockDim.x + threadIdx.x;
    if (t < LATDIM * HYPERN) {
        int d = t >> 7, h = t & 127;
        int ks = h >> 5, kl = h & 31;
        int nt = d >> 4, nn = d & 15;
        h1f[(ks * 4 + nt) * 512 + b_pos(kl, nn)] = (_Float16)leaky05(acc[t]);
    }
}

// ee'[e,d] = leaky( sum_h hyper[e,h]*H1[d,h] ) + ee[e,d]; writes f32 + swizzled f16
__global__ void k_gemm2(const float* __restrict__ hyp, const _Float16* __restrict__ h1f,
                        float* __restrict__ eeF, _Float16* __restrict__ eeHf) {
    __shared__ __align__(32) _Float16 sB[8192]; // 16 B-frags (16 KB)

    const int tid  = threadIdx.x;
    const int lane = tid & 31;
    const int w    = tid >> 5;

#pragma unroll
    for (int i = 0; i < 4; ++i)
        async_copy16(sB + (tid + 256 * i) * 8, h1f + (tid + 256 * i) * 8);
    async_wait0();
    __syncthreads();

    const int ntile = NEDGE / 16; // 6250
    const int t = blockIdx.x * 8 + w;
    if (t < ntile) { // wave-uniform: EXEC all-ones inside
        const int e0 = t * 16;
        v8f c0 = {}, c1 = {}, c2 = {}, c3 = {};
#pragma unroll
        for (int ks = 0; ks < 4; ++ks) { // K = 128 in 4 steps of 32
            v16h a = frag_a_f32(hyp + (size_t)e0 * HYPERN + ks * 32, HYPERN, lane);
            c0 = wmma_f16f32(a, ((const v16h*)sB)[(ks * 4 + 0) * 32 + lane], c0);
            c1 = wmma_f16f32(a, ((const v16h*)sB)[(ks * 4 + 1) * 32 + lane], c1);
            c2 = wmma_f16f32(a, ((const v16h*)sB)[(ks * 4 + 2) * 32 + lane], c2);
            c3 = wmma_f16f32(a, ((const v16h*)sB)[(ks * 4 + 3) * 32 + lane], c3);
        }
        const int nlo = lane & 15;
        const int mh  = (lane >> 4) << 3;
        const int slab = e0 >> 5, kbase = e0 & 31;
#pragma unroll
        for (int r = 0; r < 8; ++r) {
            const int e = e0 + mh + r;
            const int k = kbase + mh + r;
            _Float16* eh = eeHf + slab * 2048;
#pragma unroll
            for (int nt = 0; nt < 4; ++nt) {
                float x = (nt == 0 ? c0[r] : nt == 1 ? c1[r] : nt == 2 ? c2[r] : c3[r]);
                const int d = nt * 16 + nlo;
                const int idx = e * LATDIM + d;
                float out = leaky05(x) + eeF[idx];
                eeF[idx] = out;
                eh[(d >> 4) * 512 + a_pos(d & 15, k)] = (_Float16)out;
            }
        }
    }
}

// scatter: out[src[e],:] += ee[e,:] ; out[USER + tgt[e],:] += ee[e,:]
__global__ void k_scatter(const float* __restrict__ eeF,
                          const int* __restrict__ src, const int* __restrict__ tgt,
                          float* __restrict__ out) {
    int t = blockIdx.x * blockDim.x + threadIdx.x;
    if (t < NEDGE * LATDIM) {
        int e = t >> 6, d = t & 63;
        float x = eeF[t];
        atomicAdd(out + src[e] * LATDIM + d, x);
        atomicAdd(out + (NUSER + tgt[e]) * LATDIM + d, x);
    }
}

// ---------------- host launcher ----------------
extern "C" void kernel_launch(void* const* d_in, const int* in_sizes, int n_in,
                              void* d_out, int out_size, void* d_ws, size_t ws_size,
                              hipStream_t stream) {
    (void)in_sizes; (void)n_in; (void)out_size; (void)ws_size;
    const float* uE  = (const float*)d_in[0];
    const float* iE  = (const float*)d_in[1];
    const float* eT  = (const float*)d_in[2];
    const float* hyp = (const float*)d_in[3];
    const int*   src = (const int*)d_in[4];
    const int*   tgt = (const int*)d_in[5];
    const int*   val = (const int*)d_in[6];
    float* out = (float*)d_out;

    char* ws = (char*)d_ws;
    float*    eeF  = (float*)ws;                         // 25,600,000 B  (f32 [e][d])
    _Float16* eeHf = (_Float16*)(ws + 25600000);         // 12,800,000 B  (A-frag slabs)
    _Float16* hyB  = (_Float16*)(ws + 38400000);         // 25,600,000 B  (B-frag slabs)
    float*    acc  = (float*)(ws + 64000000);            //     32,768 B
    _Float16* h1f  = (_Float16*)(ws + 64032768);         //     16,384 B  (B-frags)

    const int OUTN = 2 * NUSER * LATDIM; // 12,800,000
    k_zero_f32<<<(OUTN + 255) / 256, 256, 0, stream>>>(out, OUTN);

    const int HN = NEDGE * HYPERN;
    k_cvt_hyB<<<(HN + 255) / 256, 256, 0, stream>>>(hyp, hyB, HN);

    const size_t lds_init = (size_t)(NTYPE * LATDIM * LATDIM + 8 * LATDIM) * sizeof(float);
    k_edge_init<<<1024, 256, lds_init, stream>>>(uE, iE, eT, src, tgt, val, eeF, eeHf);

    for (int layer = 0; layer < 2; ++layer) {
        k_zero_f32<<<32, 256, 0, stream>>>(acc, LATDIM * HYPERN);
        k_gemm1_accum<<<304, 256, 0, stream>>>(eeHf, hyB, acc);
        k_act_h1<<<32, 256, 0, stream>>>(acc, h1f);
        k_gemm2<<<(NEDGE / 16 + 7) / 8, 256, 0, stream>>>(hyp, h1f, eeF, eeHf);
    }

    const int SN = NEDGE * LATDIM;
    k_scatter<<<(SN + 255) / 256, 256, 0, stream>>>(eeF, src, tgt, out);
}
